// MultiHeadSelfAttention_48017734369357
// MI455X (gfx1250) — matrix operand order repaired and checked by reading, not compiled, not run on hardware
//
#include <hip/hip_runtime.h>

// ---------------- problem constants ----------------
constexpr int BATCH = 2;
constexpr int SEQ   = 2048;
constexpr int HS    = 1024;   // hidden size
constexpr int NHEAD = 16;
constexpr int HDIM  = 64;     // head dim
constexpr float SCALE = 0.125f;  // 1/sqrt(64)

// ---------------- vector types ----------------
typedef __attribute__((ext_vector_type(16))) __bf16 bf16x16;
typedef __attribute__((ext_vector_type(8)))  __bf16 bf16x8;
typedef __attribute__((ext_vector_type(4)))  __bf16 bf16x4;
typedef __attribute__((ext_vector_type(8)))  float  f32x8;

union Frag {
    bf16x16 v;
    bf16x8  h[2];
};

__device__ __forceinline__ f32x8 wmma_bf16(const Frag& a, const Frag& b, f32x8 c) {
    return __builtin_amdgcn_wmma_f32_16x16x32_bf16(
        /*neg_a=*/false, a.v, /*neg_b=*/false, b.v,
        /*c_mod=*/(short)0, c, /*reuse_a=*/false, /*reuse_b=*/false);
}

// ---- CDNA5 async global->LDS copy (ASYNCcnt-tracked, no VGPR round-trip) ----
__device__ __forceinline__ uint32_t lds_off(const void* p) {
    // flat->LDS aperture mapping uses addr[31:0] directly as the LDS address
    return (uint32_t)(uintptr_t)p;
}
__device__ __forceinline__ void async_copy16(uint32_t dst_lds, const void* src) {
    asm volatile("global_load_async_to_lds_b128 %0, %1, off"
                 :: "v"(dst_lds), "v"((uint64_t)(uintptr_t)src)
                 : "memory");
}
__device__ __forceinline__ void wait_async0() {
    asm volatile("s_wait_asynccnt 0x0" ::: "memory");
}

// ---------------- stage 0: fp32 -> bf16 conversions ----------------
__global__ __launch_bounds__(256) void cvt_x_kernel(const float* __restrict__ X,
                                                    __bf16* __restrict__ Xb) {
    int i = (blockIdx.x * 256 + threadIdx.x) * 4;
    float4 f = *(const float4*)(X + i);
    bf16x4 o;
    o[0] = (__bf16)f.x; o[1] = (__bf16)f.y; o[2] = (__bf16)f.z; o[3] = (__bf16)f.w;
    *(bf16x4*)(Xb + i) = o;
}

// W[k][n] fp32 -> Wt[n][k] bf16, LDS-tiled 32x32 transpose
__global__ __launch_bounds__(256) void cvt_wt_kernel(const float* __restrict__ W,
                                                     __bf16* __restrict__ Wt) {
    __shared__ float tile[32][33];
    const int tx = threadIdx.x & 31, ty = threadIdx.x >> 5; // 32x8
    const int n0 = blockIdx.x * 32, k0 = blockIdx.y * 32;
#pragma unroll
    for (int j = 0; j < 32; j += 8)
        tile[ty + j][tx] = W[(size_t)(k0 + ty + j) * HS + n0 + tx];
    __syncthreads();
#pragma unroll
    for (int j = 0; j < 32; j += 8)
        Wt[(size_t)(n0 + ty + j) * HS + k0 + tx] = (__bf16)tile[tx][ty + j];
}

// ---------------- shared GEMM core (async double-buffered) ----------------
// C[128 x 64] per 256-thread WG. A row-major [M][K=1024] bf16, Bt [N][K] bf16.
// Wave w owns rows w*16..w*16+15, 4 accumulators cover 64 cols.
constexpr int AS = 40;  // padded LDS row stride (elements) for A tile (128x32)
constexpr int BS = 40;  // padded LDS row stride for B tile (64x32)
constexpr int ABUF = 128 * AS;
constexpr int BBUF = 64 * BS;

__device__ __forceinline__ void gemm_core(const __bf16* __restrict__ A,
                                          const __bf16* __restrict__ Bt,
                                          int m0, int n0, f32x8 acc[4],
                                          __bf16* As, __bf16* Bs) {
    const int t = threadIdx.x;
    const int lane = t & 31, wave = t >> 5;
    const int lo = lane & 15, hi = lane >> 4;
    const int ar = t >> 1, ac = (t & 1) << 4;          // A stage: 128 rows x 2 halves
    const int br = t >> 2, bc = (t & 3) << 3;          // B stage: 64 rows x 4 segs

    const __bf16* gA0 = A + (size_t)(m0 + ar) * HS + ac;
    const __bf16* gB0 = Bt + (size_t)(n0 + br) * HS + bc;

    auto stage = [&](int k0, int buf) {
        __bf16* Ab = As + buf * ABUF;
        __bf16* Bb = Bs + buf * BBUF;
        async_copy16(lds_off(&Ab[ar * AS + ac]),     gA0 + k0);
        async_copy16(lds_off(&Ab[ar * AS + ac + 8]), gA0 + k0 + 8);
        async_copy16(lds_off(&Bb[br * BS + bc]),     gB0 + k0);
    };

    stage(0, 0);
    for (int it = 0; it < HS / 32; ++it) {
        wait_async0();          // our async fills of buf[it&1] are done
        __syncthreads();        // everyone's fills done; prev buf free to overwrite
        if (it + 1 < HS / 32) stage((it + 1) * 32, (it + 1) & 1);

        const __bf16* Ab = As + (it & 1) * ABUF;
        const __bf16* Bb = Bs + (it & 1) * BBUF;
        Frag a, b[4];
        a.h[0] = *(const bf16x8*)&Ab[(wave * 16 + lo) * AS + hi * 8];
        a.h[1] = *(const bf16x8*)&Ab[(wave * 16 + lo) * AS + 16 + hi * 8];
#pragma unroll
        for (int nt = 0; nt < 4; ++nt) {
            b[nt].h[0] = *(const bf16x8*)&Bb[(nt * 16 + lo) * BS + hi * 8];
            b[nt].h[1] = *(const bf16x8*)&Bb[(nt * 16 + lo) * BS + 16 + hi * 8];
        }
#pragma unroll
        for (int nt = 0; nt < 4; ++nt)
            acc[nt] = wmma_bf16(a, b[nt], acc[nt]);
    }
}

// ---------------- stage 1: fused QKV projection ----------------
// grid (M/128=32, N/64=16, 3). Output layout [B,NH,S,HD] bf16 per z.
__global__ __launch_bounds__(256) void qkv_gemm_kernel(
    const __bf16* __restrict__ Xb, const __bf16* __restrict__ WtAll,
    const float* __restrict__ bq, const float* __restrict__ bk,
    const float* __restrict__ bv, __bf16* __restrict__ QKV) {
    __shared__ __align__(16) __bf16 As[2 * ABUF];
    __shared__ __align__(16) __bf16 Bs[2 * BBUF];

    const int z = blockIdx.z;
    const int m0 = blockIdx.x * 128, n0 = blockIdx.y * 64;
    const float* bias = (z == 0) ? bq : ((z == 1) ? bk : bv);
    __bf16* out = QKV + (size_t)z * ((size_t)BATCH * NHEAD * SEQ * HDIM);
    const __bf16* Wt = WtAll + (size_t)z * HS * HS;

    f32x8 zero8 = {0.f, 0.f, 0.f, 0.f, 0.f, 0.f, 0.f, 0.f};
    f32x8 acc[4] = {zero8, zero8, zero8, zero8};
    gemm_core(Xb, Wt, m0, n0, acc, As, Bs);

    const int lane = threadIdx.x & 31, wave = threadIdx.x >> 5;
    const int lo = lane & 15, hi = lane >> 4;
#pragma unroll
    for (int nt = 0; nt < 4; ++nt) {
        int gn = n0 + nt * 16 + lo;
        float bb = bias[gn];
        int hh = gn >> 6, d = gn & 63;
#pragma unroll
        for (int r = 0; r < 8; ++r) {
            int gm = m0 + wave * 16 + r + hi * 8;   // = b*SEQ + s
            int b_ = gm >> 11, s_ = gm & (SEQ - 1);
            out[(((size_t)b_ * NHEAD + hh) * SEQ + s_) * HDIM + d] =
                (__bf16)(acc[nt][r] + bb);
        }
    }
}

// ---------------- stage 2: flash attention (async double-buffered K/V) ----------------
// grid (SEQ/128=16, BATCH*NHEAD=32), 256 threads. Wave w: 16 q rows.
constexpr int KS = 72;  // Ks [32 keys][64 d] padded
constexpr int VS = 40;  // Vt [64 d][32 keys] padded
constexpr int PS = 40;  // per-wave P [16 q][32 keys] padded
constexpr int KBUF = 32 * KS;
constexpr int VBUF = 64 * VS;

__global__ __launch_bounds__(256) void attn_kernel(
    const __bf16* __restrict__ Q, const __bf16* __restrict__ K,
    const __bf16* __restrict__ V, const int* __restrict__ amask,
    __bf16* __restrict__ Ctx) {
    __shared__ __align__(16) __bf16 Ks[2 * KBUF];
    __shared__ __align__(16) __bf16 Vt[2 * VBUF];
    __shared__ __align__(16) __bf16 Ps[8 * 16 * PS];

    const int t = threadIdx.x, lane = t & 31, wave = t >> 5;
    const int lo = lane & 15, hi = lane >> 4;
    const int bh = blockIdx.y;
    const int b_ = bh >> 4, h_ = bh & 15;
    const int qbase = blockIdx.x * 128;
    const size_t base = (size_t)bh * SEQ * HDIM;
    const __bf16* Qp = Q + base;
    const __bf16* Kp = K + base;
    const __bf16* Vp = V + base;
    __bf16* Pw = Ps + wave * 16 * PS;

    // Q tile A-fragments (16 rows x 64) held in registers for the whole loop
    Frag aq[2];
#pragma unroll
    for (int kk = 0; kk < 2; ++kk) {
        const __bf16* qrow = Qp + (size_t)(qbase + wave * 16 + lo) * HDIM + kk * 32;
        aq[kk].h[0] = *(const bf16x8*)(qrow + hi * 8);
        aq[kk].h[1] = *(const bf16x8*)(qrow + 16 + hi * 8);
    }

    f32x8 zero8 = {0.f, 0.f, 0.f, 0.f, 0.f, 0.f, 0.f, 0.f};
    f32x8 o[4] = {zero8, zero8, zero8, zero8};
    float rowm[8], rowl[8];
#pragma unroll
    for (int r = 0; r < 8; ++r) { rowm[r] = -3.0e38f; rowl[r] = 0.f; }

    const int kr = t >> 3, seg = (t & 7) * 8;   // staging: 32 keys x 8 segs

    // K block: async copy straight into LDS; V block: transpose via VGPRs
    auto stage = [&](int kb, int buf) {
        async_copy16(lds_off(&Ks[buf * KBUF + kr * KS + seg]),
                     Kp + (size_t)(kb + kr) * HDIM + seg);
        bf16x8 vv = *(const bf16x8*)(Vp + (size_t)(kb + kr) * HDIM + seg);
        __bf16* Vb = Vt + buf * VBUF;
#pragma unroll
        for (int j = 0; j < 8; ++j) Vb[(seg + j) * VS + kr] = vv[j];
    };

    stage(0, 0);
    for (int it = 0; it < SEQ / 32; ++it) {
        const int kb = it * 32;
        wait_async0();
        __syncthreads();
        if (it + 1 < SEQ / 32) stage(kb + 32, (it + 1) & 1);
        const __bf16* Kb = Ks + (it & 1) * KBUF;
        const __bf16* Vb = Vt + (it & 1) * VBUF;

        // scores = Q @ K^T : two 16x16 subtiles (keys lo, keys 16+lo)
        f32x8 s0 = zero8, s1 = zero8;
        Frag bk0[2], bk1[2];
#pragma unroll
        for (int kk = 0; kk < 2; ++kk) {
            bk0[kk].h[0] = *(const bf16x8*)&Kb[lo * KS + kk * 32 + hi * 8];
            bk0[kk].h[1] = *(const bf16x8*)&Kb[lo * KS + kk * 32 + 16 + hi * 8];
            bk1[kk].h[0] = *(const bf16x8*)&Kb[(16 + lo) * KS + kk * 32 + hi * 8];
            bk1[kk].h[1] = *(const bf16x8*)&Kb[(16 + lo) * KS + kk * 32 + 16 + hi * 8];
        }
#pragma unroll
        for (int kk = 0; kk < 2; ++kk) {
            s0 = wmma_bf16(aq[kk], bk0[kk], s0);
            s1 = wmma_bf16(aq[kk], bk1[kk], s1);
        }

        const float madd0 = -10000.0f * (float)amask[b_ * SEQ + kb + lo];
        const float madd1 = -10000.0f * (float)amask[b_ * SEQ + kb + 16 + lo];

        float mloc[8];
#pragma unroll
        for (int r = 0; r < 8; ++r) {
            s0[r] = s0[r] * SCALE + madd0;
            s1[r] = s1[r] * SCALE + madd1;
            mloc[r] = fmaxf(s0[r], s1[r]);
        }
        // half-wave (16-lane) row-max: xor masks 1..8 stay within each half
#pragma unroll
        for (int off = 8; off >= 1; off >>= 1)
#pragma unroll
            for (int r = 0; r < 8; ++r)
                mloc[r] = fmaxf(mloc[r], __shfl_xor(mloc[r], off, 32));

        float al[8], ps[8];
#pragma unroll
        for (int r = 0; r < 8; ++r) {
            float mnew = fmaxf(rowm[r], mloc[r]);
            al[r] = __expf(rowm[r] - mnew);
            rowm[r] = mnew;
            s0[r] = __expf(s0[r] - mnew);
            s1[r] = __expf(s1[r] - mnew);
            ps[r] = s0[r] + s1[r];
        }
#pragma unroll
        for (int off = 8; off >= 1; off >>= 1)
#pragma unroll
            for (int r = 0; r < 8; ++r) ps[r] += __shfl_xor(ps[r], off, 32);
#pragma unroll
        for (int r = 0; r < 8; ++r) rowl[r] = rowl[r] * al[r] + ps[r];
#pragma unroll
        for (int nt = 0; nt < 4; ++nt)
#pragma unroll
            for (int r = 0; r < 8; ++r) o[nt][r] *= al[r];

        // P: C-layout -> LDS -> A-layout (per-wave region; in-order DS ops)
#pragma unroll
        for (int r = 0; r < 8; ++r) {
            Pw[(r + hi * 8) * PS + lo]      = (__bf16)s0[r];
            Pw[(r + hi * 8) * PS + 16 + lo] = (__bf16)s1[r];
        }
        Frag ap, bv[4];
        ap.h[0] = *(const bf16x8*)&Pw[lo * PS + hi * 8];
        ap.h[1] = *(const bf16x8*)&Pw[lo * PS + 16 + hi * 8];
#pragma unroll
        for (int nt = 0; nt < 4; ++nt) {
            bv[nt].h[0] = *(const bf16x8*)&Vb[(nt * 16 + lo) * VS + hi * 8];
            bv[nt].h[1] = *(const bf16x8*)&Vb[(nt * 16 + lo) * VS + 16 + hi * 8];
        }
#pragma unroll
        for (int nt = 0; nt < 4; ++nt)
            o[nt] = wmma_bf16(ap, bv[nt], o[nt]);
    }

    // normalize and write ctx in [B,S,NH*HD] layout (bf16)
    float inv[8];
#pragma unroll
    for (int r = 0; r < 8; ++r) inv[r] = 1.0f / rowl[r];
#pragma unroll
    for (int nt = 0; nt < 4; ++nt) {
        int d = nt * 16 + lo;
#pragma unroll
        for (int r = 0; r < 8; ++r) {
            int s_ = qbase + wave * 16 + r + hi * 8;
            Ctx[((size_t)b_ * SEQ + s_) * HS + h_ * HDIM + d] =
                (__bf16)(o[nt][r] * inv[r]);
        }
    }
}

// ---------------- stage 3: output projection (fp32 out) ----------------
__global__ __launch_bounds__(256) void oproj_kernel(
    const __bf16* __restrict__ Ctx, const __bf16* __restrict__ Wot,
    const float* __restrict__ bo, float* __restrict__ Out) {
    __shared__ __align__(16) __bf16 As[2 * ABUF];
    __shared__ __align__(16) __bf16 Bs[2 * BBUF];

    const int m0 = blockIdx.x * 128, n0 = blockIdx.y * 64;
    f32x8 zero8 = {0.f, 0.f, 0.f, 0.f, 0.f, 0.f, 0.f, 0.f};
    f32x8 acc[4] = {zero8, zero8, zero8, zero8};
    gemm_core(Ctx, Wot, m0, n0, acc, As, Bs);

    const int lane = threadIdx.x & 31, wave = threadIdx.x >> 5;
    const int lo = lane & 15, hi = lane >> 4;
#pragma unroll
    for (int nt = 0; nt < 4; ++nt) {
        int gn = n0 + nt * 16 + lo;
        float bb = bo[gn];
#pragma unroll
        for (int r = 0; r < 8; ++r) {
            int gm = m0 + wave * 16 + r + hi * 8;
            Out[(size_t)gm * HS + gn] = acc[nt][r] + bb;
        }
    }
}

// ---------------- host launch ----------------
extern "C" void kernel_launch(void* const* d_in, const int* in_sizes, int n_in,
                              void* d_out, int out_size, void* d_ws, size_t ws_size,
                              hipStream_t stream) {
    const float* hs    = (const float*)d_in[0];
    const int*   amask = (const int*)d_in[1];
    const float* Wq    = (const float*)d_in[2];
    const float* bq    = (const float*)d_in[3];
    const float* Wk    = (const float*)d_in[4];
    const float* bk    = (const float*)d_in[5];
    const float* Wv    = (const float*)d_in[6];
    const float* bv    = (const float*)d_in[7];
    const float* Wo    = (const float*)d_in[8];
    const float* bo    = (const float*)d_in[9];

    const size_t NTOK = (size_t)BATCH * SEQ;           // 4096
    const size_t XN   = NTOK * HS;                     // 4M elems
    const size_t WN   = (size_t)HS * HS;               // 1M elems
    const size_t HN   = (size_t)BATCH * NHEAD * SEQ * HDIM;  // 4M elems

    __bf16* Xb  = (__bf16*)d_ws;
    __bf16* Wt  = Xb + XN;         // 4 transposed weights, contiguous
    __bf16* Qb  = Wt + 4 * WN;     // Q,K,V contiguous (qkv kernel indexes by z)
    __bf16* Kb  = Qb + HN;
    __bf16* Vb  = Kb + HN;
    __bf16* Ctx = Vb + HN;

    cvt_x_kernel<<<(int)(XN / 1024), 256, 0, stream>>>(hs, Xb);
    cvt_wt_kernel<<<dim3(32, 32), 256, 0, stream>>>(Wq, Wt);
    cvt_wt_kernel<<<dim3(32, 32), 256, 0, stream>>>(Wk, Wt + WN);
    cvt_wt_kernel<<<dim3(32, 32), 256, 0, stream>>>(Wv, Wt + 2 * WN);
    cvt_wt_kernel<<<dim3(32, 32), 256, 0, stream>>>(Wo, Wt + 3 * WN);

    qkv_gemm_kernel<<<dim3(NTOK / 128, HS / 64, 3), 256, 0, stream>>>(
        Xb, Wt, bq, bk, bv, Qb);

    attn_kernel<<<dim3(SEQ / 128, BATCH * NHEAD), 256, 0, stream>>>(
        Qb, Kb, Vb, amask, Ctx);

    oproj_kernel<<<dim3(NTOK / 128, HS / 64), 256, 0, stream>>>(
        Ctx, Wt + 3 * WN, bo, (float*)d_out);
}
